// _LSTMLayer_30253749633774
// MI455X (gfx1250) — compile-verified, matrix operand order checked by reading, not run on hardware
//
#include <hip/hip_runtime.h>
#include <hip/hip_bf16.h>
#include <math.h>

typedef __attribute__((ext_vector_type(16))) __bf16 v16bf;
typedef __attribute__((ext_vector_type(8)))  __bf16 v8bf;
typedef __attribute__((ext_vector_type(8)))  float  v8f;
typedef __attribute__((ext_vector_type(4)))  unsigned tdm_v4u;
typedef __attribute__((ext_vector_type(8)))  int      tdm_v8i;
typedef __attribute__((ext_vector_type(4)))  int      tdm_v4i;

#define T_SEQ 4096
#define HID   2048
#define G4H   8192   // 4*HID

#if __has_builtin(__builtin_amdgcn_tensor_load_to_lds) && __has_builtin(__builtin_amdgcn_s_wait_tensorcnt)
#define HAVE_TDM 1
#else
#define HAVE_TDM 0
#endif

// ---------------------------------------------------------------------------
// fp32 -> bf16 conversion (vectorized x4)
// ---------------------------------------------------------------------------
__global__ __launch_bounds__(256)
void lstm_cvt_f32_bf16(const float* __restrict__ in, __bf16* __restrict__ out, int n) {
    int i = (blockIdx.x * blockDim.x + threadIdx.x) * 4;
    if (i + 3 < n) {
        float4 v = *(const float4*)(in + i);
        out[i + 0] = (__bf16)v.x;
        out[i + 1] = (__bf16)v.y;
        out[i + 2] = (__bf16)v.z;
        out[i + 3] = (__bf16)v.w;
    }
}

// ---------------------------------------------------------------------------
// GEMM tiling parameters
// ---------------------------------------------------------------------------
#define BM 128
#define BN 64
#define BK 32
#define LDA 40   // padded LDS stride (bf16): 80B rows -> conflict-free b128 frag loads
#define LDB 40

// LDS layout (dynamic, base byte offset 0):
//   A buffers: elems [0, 5120) and [5120, 10240)     (byte 0 / 10240, 10240B each)
//   B buffers: elems [10240, 12800) and [12800,15360) (byte 20480 / 25600, 5120B each)
#define A_BUF_ELEMS (BM * LDA)           // 5120
#define B_BUF_ELEMS (BN * LDB)           // 2560
#define A_ELEM_OFF(b)  ((b) * A_BUF_ELEMS)
#define B_ELEM_OFF(b)  (2 * A_BUF_ELEMS + (b) * B_BUF_ELEMS)
#define A_BYTE_OFF(b)  ((unsigned)(A_ELEM_OFF(b) * 2))
#define DYN_LDS_BYTES  ((2 * A_BUF_ELEMS + 2 * B_BUF_ELEMS) * 2)   // 30720

// ---------------------------------------------------------------------------
// TDM: async-load one 128x32 bf16 tile of X (T_SEQ x HID row-major) into LDS,
// hardware-padding each 64B row with 16B so LDS stride == LDA (40 bf16).
// Descriptor per CDNA5 ISA ch.8 (D# groups 0/1; groups 2/3 zero for 2D).
// ---------------------------------------------------------------------------
#if HAVE_TDM
__device__ __forceinline__ void tdm_load_A_tile(const __bf16* gsrc, unsigned lds_byte_off) {
    unsigned long long ga = (unsigned long long)(uintptr_t)gsrc;
    tdm_v4u g0;
    g0[0] = 1u;                                   // count=1 valid user descriptor
    g0[1] = lds_byte_off;                         // lds_addr (bytes)
    g0[2] = (unsigned)(ga & 0xFFFFFFFFu);         // global_addr[31:0]
    g0[3] = (unsigned)((ga >> 32) & 0x01FFFFFFu)  // global_addr[56:32]
          | (2u << 30);                           // type = 2 ("image")
    tdm_v8i g1;
    g1[0] = (int)((1u << 16)      // data_size = 1 -> 2 bytes
                | (1u << 20)      // pad_enable
                | (3u << 22)      // pad_interval: 16 DWORDs (one 64B tile row)
                | (3u << 25));    // pad_amount:   4 DWORDs (16B -> stride 80B)
    g1[1] = (int)(((unsigned)HID & 0xFFFFu) << 16);    // tensor_dim0 lo16 (2048)
    g1[2] = (int)(((unsigned)T_SEQ & 0xFFFFu) << 16);  // dim0 hi=0 | tensor_dim1 lo16 (4096)
    g1[3] = (int)((unsigned)BK << 16);                 // dim1 hi=0 | tile_dim0 = 32
    g1[4] = (int)(unsigned)BM;                         // tile_dim1 = 128, tile_dim2 = 0
    g1[5] = (int)(unsigned)HID;                        // tensor_dim0_stride lo32 = 2048
    g1[6] = 0;                                         // stride hi | dim1_stride lo
    g1[7] = 0;
    tdm_v4i gz4 = {0, 0, 0, 0};
    tdm_v8i gz8 = {0, 0, 0, 0, 0, 0, 0, 0};
    __builtin_amdgcn_tensor_load_to_lds(g0, g1, gz4, gz4, gz8, 0);
}
#endif

// ---------------------------------------------------------------------------
// Pre-pass GEMM:  Ypre[T][4H] = X[T][H] @ Wi[H][4H]   (bf16 WMMA, fp32 accum)
// Block tile 128(M) x 64(N), K-step 32, double-buffered LDS.
// A tiles staged by the Tensor Data Mover (async, s_wait_tensorcnt),
// B tiles staged manually (transposed scatter) overlapping compute.
// ---------------------------------------------------------------------------
__global__ __launch_bounds__(256)
void lstm_gemm_pre(const __bf16* __restrict__ A,   // (T_SEQ, HID)
                   const __bf16* __restrict__ B,   // (HID,  G4H)
                   float* __restrict__ C)          // (T_SEQ, G4H)
{
    extern __shared__ __bf16 smem[];

    const int tid  = threadIdx.x;
    const int wave = tid >> 5;        // 0..7
    const int lane = tid & 31;
    const int wm   = wave >> 1;       // 0..3 -> 32*wm row offset
    const int wn   = wave & 1;        // 0..1 -> 32*wn col offset
    const int l15  = lane & 15;
    const int half = lane >> 4;

    const int m0 = blockIdx.y * BM;
    const int n0 = blockIdx.x * BN;

    // B staging thread mapping: read 8 contiguous n, scatter to sB[n][k]
    const int bkk = tid >> 3;            // 0..31
    const int bnn = (tid & 7) * 8;       // 0..56

    v8f acc[2][2] = {};

#if !HAVE_TDM
    auto stage_A = [&](int k0, int bufb) {
        for (int cch = tid; cch < (BM * BK) / 8; cch += 256) {
            int r  = cch >> 2;
            int kk = (cch & 3) * 8;
            v8bf v = *(const v8bf*)(A + (size_t)(m0 + r) * HID + k0 + kk);
            *(v8bf*)(smem + A_ELEM_OFF(bufb) + r * LDA + kk) = v;
        }
    };
#endif

    // ---------------- prologue: stage tile k=0 into buffer 0 ----------------
#if HAVE_TDM
    if (wave == 0)
        tdm_load_A_tile(A + (size_t)m0 * HID, A_BYTE_OFF(0));
#else
    stage_A(0, 0);
#endif
    {
        v8bf v = *(const v8bf*)(B + (size_t)bkk * G4H + n0 + bnn);
        #pragma unroll
        for (int i = 0; i < 8; ++i)
            smem[B_ELEM_OFF(0) + (bnn + i) * LDB + bkk] = v[i];
    }
#if HAVE_TDM
    if (wave == 0) __builtin_amdgcn_s_wait_tensorcnt(0);
#endif
    __syncthreads();

    // ---------------- main loop, double buffered ----------------
    int buf = 0;
    const int NK = HID / BK;   // 64
    for (int kt = 0; kt < NK; ++kt) {
        const int nxt  = buf ^ 1;
        const bool more = (kt + 1 < NK);
        const int knext = (kt + 1) * BK;

        v8bf bnext;
        if (more) {
#if HAVE_TDM
            if (wave == 0)
                tdm_load_A_tile(A + (size_t)m0 * HID + knext, A_BYTE_OFF(nxt));
#else
            stage_A(knext, nxt);
#endif
            bnext = *(const v8bf*)(B + (size_t)(knext + bkk) * G4H + n0 + bnn);
        }

        // ---- fragments + WMMA on current buffers ----
        const __bf16* sA = smem + A_ELEM_OFF(buf);
        const __bf16* sB = smem + B_ELEM_OFF(buf);
        #pragma unroll
        for (int mi = 0; mi < 2; ++mi) {
            const __bf16* ap = sA + (32 * wm + 16 * mi + l15) * LDA + half * 8;
            union { v16bf v; v8bf h[2]; } af;
            af.h[0] = *(const v8bf*)(ap);
            af.h[1] = *(const v8bf*)(ap + 16);
            #pragma unroll
            for (int ni = 0; ni < 2; ++ni) {
                const __bf16* bp = sB + (32 * wn + 16 * ni + l15) * LDB + half * 8;
                union { v16bf v; v8bf h[2]; } bfr;
                bfr.h[0] = *(const v8bf*)(bp);
                bfr.h[1] = *(const v8bf*)(bp + 16);
                acc[mi][ni] = __builtin_amdgcn_wmma_f32_16x16x32_bf16(
                    false, af.v, false, bfr.v, (short)0, acc[mi][ni], false, false);
            }
        }

        if (more) {
            #pragma unroll
            for (int i = 0; i < 8; ++i)
                smem[B_ELEM_OFF(nxt) + (bnn + i) * LDB + bkk] = bnext[i];
#if HAVE_TDM
            if (wave == 0) __builtin_amdgcn_s_wait_tensorcnt(0);
#endif
        }
        __syncthreads();
        buf = nxt;
    }

    // ---- store D: VGPR r, lanes0-15 -> M=r, lanes16-31 -> M=r+8 ----
    #pragma unroll
    for (int mi = 0; mi < 2; ++mi)
        #pragma unroll
        for (int ni = 0; ni < 2; ++ni) {
            int n = n0 + 32 * wn + 16 * ni + l15;
            int mbase = m0 + 32 * wm + 16 * mi + 8 * half;
            #pragma unroll
            for (int r = 0; r < 8; ++r)
                C[(size_t)(mbase + r) * G4H + n] = acc[mi][ni][r];
        }
}

// ---------------------------------------------------------------------------
// init: carry state, gate accumulator, barrier counters
// ---------------------------------------------------------------------------
__global__ __launch_bounds__(256)
void lstm_init(const float* __restrict__ c0, const float* __restrict__ h0,
               float* __restrict__ c, __bf16* __restrict__ hb,
               float* __restrict__ acc, unsigned* __restrict__ bar)
{
    int i = blockIdx.x * blockDim.x + threadIdx.x;
    if (i < HID) { c[i] = c0[i]; hb[i] = (__bf16)h0[i]; }
    if (i < G4H) acc[i] = 0.0f;
    if (i < 4)   bar[i] = 0u;
}

// ---------------------------------------------------------------------------
// device-wide split barrier (persistent kernel)
// ---------------------------------------------------------------------------
__device__ __forceinline__ void grid_sync(unsigned* cnt, unsigned* gen, unsigned nb) {
    __syncthreads();
    if (threadIdx.x == 0) {
        unsigned g = __hip_atomic_load(gen, __ATOMIC_RELAXED, __HIP_MEMORY_SCOPE_AGENT);
        if (__hip_atomic_fetch_add(cnt, 1u, __ATOMIC_ACQ_REL, __HIP_MEMORY_SCOPE_AGENT) == nb - 1u) {
            __hip_atomic_store(cnt, 0u, __ATOMIC_RELAXED, __HIP_MEMORY_SCOPE_AGENT);
            __hip_atomic_fetch_add(gen, 1u, __ATOMIC_ACQ_REL, __HIP_MEMORY_SCOPE_AGENT);
        } else {
            while (__hip_atomic_load(gen, __ATOMIC_ACQUIRE, __HIP_MEMORY_SCOPE_AGENT) == g)
                __builtin_amdgcn_s_sleep(2);
        }
    }
    __syncthreads();
}

__device__ __forceinline__ float sigmoidf_(float x) {
    return 1.0f / (1.0f + __expf(-x));
}

// ---------------------------------------------------------------------------
// Persistent recurrent kernel. Grid (32, 8) x 256 threads.
//  gx selects 256 gate outputs, gy selects a 256-wide K chunk.
//  Per step: partial GEMV -> f32 global atomics -> barrier -> gate math.
//  Wh (bf16, 32MB) is L2-resident: lanes read consecutive columns of a Wh
//  row (coalesced 64B/wave); global_prefetch_b8 runs ahead of the row walk.
// ---------------------------------------------------------------------------
#define KCHUNK 256

__global__ __launch_bounds__(256)
void lstm_recurrent(const float* __restrict__ Ypre,  // (T,4H)
                    const __bf16* __restrict__ Wh,   // (H,4H) bf16
                    const float* __restrict__ b,     // (4H)
                    float* __restrict__ acc,         // (4H) partial gate sums
                    float* __restrict__ c,           // (H) cell state
                    __bf16* __restrict__ hb,         // (H) hidden, bf16
                    float* __restrict__ out,         // [c_f | h_f | ys]
                    unsigned* __restrict__ bar,      // [cnt, gen]
                    unsigned nblocks)
{
    const int tid   = threadIdx.x;
    const int gx    = blockIdx.x;          // 0..31
    const int gy    = blockIdx.y;          // 0..7
    const int j     = gx * 256 + tid;      // gate output index 0..8191
    const int kbase = gy * KCHUNK;

    __shared__ float sh[KCHUNK];

    for (int t = 0; t < T_SEQ; ++t) {
        // h(t-1) published + acc reset completed
        grid_sync(bar, bar + 1, nblocks);

        sh[tid] = (float)hb[kbase + tid];
        __syncthreads();

        const __bf16* wp = Wh + (size_t)kbase * G4H + j;
        float partial = 0.0f;
        for (int k = 0; k < KCHUNK; k += 16) {
            __builtin_prefetch(wp + (size_t)(k + 32) * G4H, 0, 0);  // run ahead in L2
            #pragma unroll
            for (int kk = 0; kk < 16; ++kk)
                partial += sh[k + kk] * (float)wp[(size_t)(k + kk) * G4H];
        }

        __hip_atomic_fetch_add(&acc[j], partial, __ATOMIC_RELAXED, __HIP_MEMORY_SCOPE_AGENT);

        // all partials landed
        grid_sync(bar, bar + 1, nblocks);

        if (gy == 0 && gx < 8) {
            int n = gx * 256 + tid;        // hidden unit 0..2047
            const float* yp = Ypre + (size_t)t * G4H;
            float yi = acc[n]           + yp[n]           + b[n];
            float yf = acc[n + HID]     + yp[n + HID]     + b[n + HID];
            float yg = acc[n + 2 * HID] + yp[n + 2 * HID] + b[n + 2 * HID];
            float yo = acc[n + 3 * HID] + yp[n + 3 * HID] + b[n + 3 * HID];

            float ig = sigmoidf_(yi);
            float fg = sigmoidf_(yf);
            float gg = tanhf(yg);
            float og = sigmoidf_(yo);

            float cn = fg * c[n] + ig * gg;
            float hn = og * tanhf(cn);

            c[n]  = cn;
            hb[n] = (__bf16)hn;
            out[2 * HID + (size_t)t * HID + n] = hn;
            if (t == T_SEQ - 1) { out[n] = cn; out[HID + n] = hn; }

            // reset accumulator for next step (published by next grid_sync)
            acc[n] = 0.0f; acc[n + HID] = 0.0f;
            acc[n + 2 * HID] = 0.0f; acc[n + 3 * HID] = 0.0f;
        }
    }
}

// ---------------------------------------------------------------------------
// launch
// ---------------------------------------------------------------------------
static inline size_t align256(size_t x) { return (x + 255) & ~(size_t)255; }

extern "C" void kernel_launch(void* const* d_in, const int* in_sizes, int n_in,
                              void* d_out, int out_size, void* d_ws, size_t ws_size,
                              hipStream_t stream) {
    const float* x  = (const float*)d_in[0];   // (T,H)
    const float* c0 = (const float*)d_in[1];   // (H)
    const float* h0 = (const float*)d_in[2];   // (H)
    const float* Wi = (const float*)d_in[3];   // (H,4H)
    const float* Wh = (const float*)d_in[4];   // (H,4H)
    const float* b  = (const float*)d_in[5];   // (4H)
    float* out = (float*)d_out;                // [c_f(H) | h_f(H) | ys(T*H)]

    char* ws = (char*)d_ws;
    size_t off = 0;
    __bf16* xb  = (__bf16*)(ws + off); off = align256(off + (size_t)T_SEQ * HID * 2);
    __bf16* wib = (__bf16*)(ws + off); off = align256(off + (size_t)HID * G4H * 2);
    __bf16* whb = (__bf16*)(ws + off); off = align256(off + (size_t)HID * G4H * 2);
    float*  ypre = (float*)(ws + off); off = align256(off + (size_t)T_SEQ * G4H * 4);
    float*  acc  = (float*)(ws + off); off = align256(off + (size_t)G4H * 4);
    float*  cbuf = (float*)(ws + off); off = align256(off + (size_t)HID * 4);
    __bf16* hb   = (__bf16*)(ws + off); off = align256(off + (size_t)HID * 2);
    unsigned* bar = (unsigned*)(ws + off); off = align256(off + 256);
    (void)ws_size; (void)n_in; (void)in_sizes; (void)out_size;

    // 1) fp32 -> bf16 conversions
    {
        int n = T_SEQ * HID;
        lstm_cvt_f32_bf16<<<n / (256 * 4), 256, 0, stream>>>(x, xb, n);
        n = HID * G4H;
        lstm_cvt_f32_bf16<<<n / (256 * 4), 256, 0, stream>>>(Wi, wib, n);
        lstm_cvt_f32_bf16<<<n / (256 * 4), 256, 0, stream>>>(Wh, whb, n);
    }

    // 2) Ypre = X @ Wi  (WMMA bf16, TDM-staged A tiles)
    {
        dim3 grid(G4H / BN, T_SEQ / BM);   // (128, 32)
        lstm_gemm_pre<<<grid, 256, DYN_LDS_BYTES, stream>>>(xb, wib, ypre);
    }

    // 3) init state + barrier
    lstm_init<<<G4H / 256, 256, 0, stream>>>(c0, h0, cbuf, hb, acc, bar);

    // 4) persistent recurrence
    {
        dim3 grid(G4H / 256, HID / KCHUNK);  // (32, 8) = 256 blocks
        lstm_recurrent<<<grid, 256, 0, stream>>>(ypre, whb, b, acc, cbuf, hb,
                                                 out, bar, 32u * 8u);
    }
}
